// Attention_89678917140909
// MI455X (gfx1250) — matrix operand order repaired and checked by reading, not compiled, not run on hardware
//
#include <hip/hip_runtime.h>
#include <hip/hip_bf16.h>

// ---------------------------------------------------------------------------
// GQA attention for MI455X (gfx1250, wave32, WMMA bf16 16x16x32, f32 accum)
// B=2 T=2048 D=2048 N=16 K=4 H=128 G=4
// Pipeline: [GEMM+RoPE -> Q,K bf16] [GEMM -> V^T bf16] [TDM-staged flash attn]
//           [GEMM -> out f32]
// ---------------------------------------------------------------------------

#define B_  2
#define T_  2048
#define D_  2048
#define NH_ 16
#define KH_ 4
#define H_  128
#define G_  4

typedef __attribute__((ext_vector_type(16))) __bf16 v16bf;
typedef __attribute__((ext_vector_type(8)))  float  v8f;
typedef __attribute__((ext_vector_type(4)))  unsigned int v4u;
typedef __attribute__((ext_vector_type(8)))  int    v8i;
typedef __attribute__((ext_vector_type(4)))  int    v4i;

union FragA { v16bf v; unsigned int u[8]; };
union FragC { v8f   v; float        f[8]; };

__device__ __forceinline__ unsigned short f2bf(float x) {
  unsigned int u = __float_as_uint(x);
  u += 0x7FFFu + ((u >> 16) & 1u);          // round-to-nearest-even
  return (unsigned short)(u >> 16);
}

// ---------------------------------------------------------------------------
// Tensor Data Mover: 2D bf16 tile (tile_d0 contiguous elems x tile_d1 rows,
// row stride stride0 elems) global -> LDS, with 1 dword of LDS padding every
// pad_interval_code (3 => 16 dwords, 5 => 64 dwords) to kill bank conflicts.
// D# packing per CDNA5 ISA.
// ---------------------------------------------------------------------------
#if __has_builtin(__builtin_amdgcn_tensor_load_to_lds)
#define USE_TDM 1
#else
#define USE_TDM 0
#endif

#if USE_TDM
__device__ __forceinline__ void tdm_load_2d_bf16(unsigned lds_addr, const void* gptr,
                                                 unsigned tile_d0, unsigned tile_d1,
                                                 unsigned long long stride0,
                                                 unsigned pad_interval_code)
{
  unsigned long long ga = (unsigned long long)gptr;
  v4u g0;
  g0[0] = 1u;                                             // count=1, user mode
  g0[1] = lds_addr;                                       // LDS byte address
  g0[2] = (unsigned)(ga & 0xffffffffu);                   // global_addr[31:0]
  g0[3] = (unsigned)((ga >> 32) & 0x1ffffffu) | (2u << 30); // addr[56:32] | type=2
  v8i g1;
  // data_size=2B (1<<16), pad_enable (1<<20), pad_interval code, pad_amount=0 (1 dword)
  g1[0] = (int)((1u << 16) | (1u << 20) | (pad_interval_code << 22));
  g1[1] = (int)((tile_d0 & 0xffffu) << 16);               // tensor_dim0[15:0]
  g1[2] = (int)(((tile_d0 >> 16) & 0xffffu) | ((tile_d1 & 0xffffu) << 16)); // td0 hi | tensor_dim1 lo
  g1[3] = (int)(((tile_d1 >> 16) & 0xffffu) | ((tile_d0 & 0xffffu) << 16)); // td1 hi | tile_dim0
  g1[4] = (int)(tile_d1 & 0xffffu);                       // tile_dim1 | tile_dim2=0
  g1[5] = (int)(unsigned)(stride0 & 0xffffffffu);         // tensor_dim0_stride[31:0]
  g1[6] = (int)(unsigned)((stride0 >> 32) & 0xffffu);     // stride[47:32] | dim1_stride lo = 0
  g1[7] = 0;
  v4i z4 = {0, 0, 0, 0};
#if __clang_major__ >= 23
  v8i z8 = {0, 0, 0, 0, 0, 0, 0, 0};
  __builtin_amdgcn_tensor_load_to_lds(g0, g1, z4, z4, z8, 0);
#else
  __builtin_amdgcn_tensor_load_to_lds(g0, g1, z4, z4, 0);
#endif
}

__device__ __forceinline__ void wait_tensor0() {
#if __has_builtin(__builtin_amdgcn_s_wait_tensorcnt)
  __builtin_amdgcn_s_wait_tensorcnt(0);
#else
  asm volatile("s_wait_tensorcnt 0x0" ::: "memory");
#endif
}
#endif // USE_TDM

// ---------------------------------------------------------------------------
// Generic 128x128-tile GEMM:  C[M,Ncols] = A[M,KD] * Bw[KD,Ncols]
//   A f32 (converted) or bf16; Bw f32 (converted).
//   do_rope: rotate (c, c+64) pairs in-register, emit bf16.
//   out_transpose: emit bf16 column-major (for V^T).
// ---------------------------------------------------------------------------
__global__ __launch_bounds__(256)
void gemm_bf16_wmma_kernel(const float* __restrict__ Af32,
                           const unsigned short* __restrict__ Abf,
                           const float* __restrict__ Bw,
                           unsigned short* __restrict__ Obf,
                           float* __restrict__ Of32,
                           const int* __restrict__ positions,
                           int KD, int Ncols, int Mtot,
                           int do_rope, int out_transpose)
{
  __shared__ unsigned short As[128][34];   // [row][k]  (+2 pad)
  __shared__ unsigned short Bs[128][34];   // [n][k]    (B transposed)

  const int tid  = threadIdx.x;
  const int lane = tid & 31;
  const int wid  = tid >> 5;
  const int wm   = wid & 3;
  const int wn   = wid >> 2;
  const int half = lane >> 4;
  const int lm   = lane & 15;

  const int m0 = blockIdx.x * 128;
  const int n0 = blockIdx.y * 128;

  FragC acc[2][4];
#pragma unroll
  for (int i = 0; i < 2; ++i)
#pragma unroll
    for (int j = 0; j < 4; ++j)
#pragma unroll
      for (int r = 0; r < 8; ++r) acc[i][j].f[r] = 0.0f;

  const int arow = tid >> 1;
  const int ac0  = (tid & 1) << 4;
  const int bkr  = tid >> 3;
  const int bc0  = (tid & 7) << 4;

  for (int kt = 0; kt < KD; kt += 32) {
    {
      long g = (long)(m0 + arow) * KD + kt + ac0;
      if (Af32) {
#pragma unroll
        for (int i = 0; i < 16; ++i) As[arow][ac0 + i] = f2bf(Af32[g + i]);
      } else {
#pragma unroll
        for (int i = 0; i < 16; ++i) As[arow][ac0 + i] = Abf[g + i];
      }
    }
    {
      long g = (long)(kt + bkr) * Ncols + n0 + bc0;
#pragma unroll
      for (int i = 0; i < 16; ++i) Bs[bc0 + i][bkr] = f2bf(Bw[g + i]);
    }
    __syncthreads();

    FragA a[2], b[4];
#pragma unroll
    for (int fm = 0; fm < 2; ++fm) {
      int row = wm * 32 + fm * 16 + lm;
#pragma unroll
      for (int j = 0; j < 8; ++j) {
        int k = ((j >= 4) ? 16 : 0) + half * 8 + ((j & 3) << 1);
        a[fm].u[j] = *(const unsigned int*)&As[row][k];
      }
    }
#pragma unroll
    for (int fn = 0; fn < 4; ++fn) {
      int cb = (fn < 2) ? (wn * 32 + fn * 16) : (64 + wn * 32 + (fn - 2) * 16);
      int nrow = cb + lm;
#pragma unroll
      for (int j = 0; j < 8; ++j) {
        int k = ((j >= 4) ? 16 : 0) + half * 8 + ((j & 3) << 1);
        b[fn].u[j] = *(const unsigned int*)&Bs[nrow][k];
      }
    }
#pragma unroll
    for (int fm = 0; fm < 2; ++fm)
#pragma unroll
      for (int fn = 0; fn < 4; ++fn)
        acc[fm][fn].v = __builtin_amdgcn_wmma_f32_16x16x32_bf16(
            false, a[fm].v, false, b[fn].v, (short)0, acc[fm][fn].v, false, false);
    __syncthreads();
  }

  if (do_rope) {
    const float ln1e4 = 9.210340371976184f;
#pragma unroll
    for (int fm = 0; fm < 2; ++fm)
#pragma unroll
      for (int fn = 0; fn < 2; ++fn) {
        int c = wn * 32 + fn * 16 + lm;                     // 0..63
        float tsinv = __expf(-((float)c * (1.0f / 64.0f)) * ln1e4);
#pragma unroll
        for (int r = 0; r < 8; ++r) {
          int row = m0 + wm * 32 + fm * 16 + r + half * 8;
          float ang = (float)positions[row] * tsinv;
          float sn, cs;
          __sincosf(ang, &sn, &cs);
          float x1 = acc[fm][fn].f[r];
          float x2 = acc[fm][fn + 2].f[r];
          long base = (long)row * Ncols + n0;
          Obf[base + c]      = f2bf(x1 * cs - x2 * sn);
          Obf[base + c + 64] = f2bf(x2 * cs + x1 * sn);
        }
      }
  } else {
#pragma unroll
    for (int fm = 0; fm < 2; ++fm)
#pragma unroll
      for (int fn = 0; fn < 4; ++fn) {
        int cb = (fn < 2) ? (wn * 32 + fn * 16) : (64 + wn * 32 + (fn - 2) * 16);
        int c = cb + lm;
#pragma unroll
        for (int r = 0; r < 8; ++r) {
          int row = m0 + wm * 32 + fm * 16 + r + half * 8;
          float v = acc[fm][fn].f[r];
          if (Of32) {
            Of32[(long)row * Ncols + n0 + c] = v;
          } else if (out_transpose) {
            Obf[(long)(n0 + c) * Mtot + row] = f2bf(v);   // column-major (V^T)
          } else {
            Obf[(long)row * Ncols + n0 + c] = f2bf(v);
          }
        }
      }
  }
}

// ---------------------------------------------------------------------------
// Flash attention: one wave per block, 16 q-rows, 32 keys per step.
// K block (32x128) and V^T block (128x32) are DMA'd into LDS by the TDM with
// per-row dword padding (bank-conflict-free fragment reads), then
// S = Q K^T (8 wmma), online softmax, O += P V (8 wmma).
// ---------------------------------------------------------------------------
#define KS_STRIDE 130   // 128 bf16 + 1 dword pad  (65 dwords/row)
#define VS_STRIDE 34    //  32 bf16 + 1 dword pad  (17 dwords/row)
#define KS_WORDS  (32 * KS_STRIDE)
#define VS_WORDS  (128 * VS_STRIDE)
#define PS_WORDS  (16 * 34)

__global__ __launch_bounds__(32)
void flash_fwd_kernel(const unsigned short* __restrict__ Qb,
                      const unsigned short* __restrict__ Kb,
                      const unsigned short* __restrict__ Vt,
                      unsigned short* __restrict__ Ab)
{
  const int qt = blockIdx.x;
  const int bn = blockIdx.y;
  const int b  = bn / NH_;
  const int n  = bn % NH_;
  const int kh = n / G_;
  const int lane = threadIdx.x;
  const int half = lane >> 4;
  const int lm   = lane & 15;

  // single static LDS blob: static LDS allocation starts at offset 0
  __shared__ __align__(16) unsigned short smem[KS_WORDS + VS_WORDS + PS_WORDS];
  unsigned short* Ks = smem;                    // [32][KS_STRIDE]
  unsigned short* Vs = smem + KS_WORDS;         // [128][VS_STRIDE]
  unsigned short* Ps = smem + KS_WORDS + VS_WORDS;
  const unsigned LDS_K = 0;
  const unsigned LDS_V = KS_WORDS * 2;

  FragA aq[4];
  {
    long base = (long)(b * T_ + qt * 16 + lm) * (NH_ * H_) + (long)n * H_;
#pragma unroll
    for (int hc = 0; hc < 4; ++hc)
#pragma unroll
      for (int j = 0; j < 8; ++j) {
        int k = ((j >= 4) ? 16 : 0) + half * 8 + ((j & 3) << 1);
        aq[hc].u[j] = *(const unsigned int*)&Qb[base + hc * 32 + k];
      }
  }

  float m_i[8], l_i[8];
  FragC o[8];
#pragma unroll
  for (int r = 0; r < 8; ++r) { m_i[r] = -1e30f; l_i[r] = 0.0f; }
#pragma unroll
  for (int f = 0; f < 8; ++f)
#pragma unroll
    for (int r = 0; r < 8; ++r) o[f].f[r] = 0.0f;

  const float scale = 0.088388347648318447f;   // 1/sqrt(128)
  const int kend = qt * 16 + 16;

  for (int kb = 0; kb < kend; kb += 32) {
    __syncthreads();   // all LDS reads of previous step complete before overwrite

    const long kbase = (long)(b * T_ + kb) * (KH_ * H_) + (long)kh * H_;
    const long vbase = (long)(kh * H_) * (B_ * T_) + (long)b * T_ + kb;
#if USE_TDM
    // K block: 32 key-rows x 128 bf16, row stride K*H elems; pad 1 dw / 64 dw
    tdm_load_2d_bf16(LDS_K, Kb + kbase, 128u, 32u,
                     (unsigned long long)(KH_ * H_), 5u);
    // V^T block: 128 h-rows x 32 bf16, row stride B*T elems; pad 1 dw / 16 dw
    tdm_load_2d_bf16(LDS_V, Vt + vbase, 32u, 128u,
                     (unsigned long long)(B_ * T_), 3u);
    wait_tensor0();
#else
    // cooperative dword staging fallback
    for (int i = lane; i < 32 * 64; i += 32) {   // K: 32 rows x 64 dwords
      int row = i >> 6, c = i & 63;
      *(unsigned int*)&Ks[row * KS_STRIDE + 2 * c] =
          *(const unsigned int*)&Kb[kbase + (long)row * (KH_ * H_) + 2 * c];
    }
    for (int i = lane; i < 128 * 16; i += 32) {  // V^T: 128 rows x 16 dwords
      int row = i >> 4, c = i & 15;
      *(unsigned int*)&Vs[row * VS_STRIDE + 2 * c] =
          *(const unsigned int*)&Vt[vbase + (long)row * (B_ * T_) + 2 * c];
    }
    __syncthreads();
#endif

    FragC s0, s1;
#pragma unroll
    for (int r = 0; r < 8; ++r) { s0.f[r] = 0.0f; s1.f[r] = 0.0f; }

#pragma unroll
    for (int hc = 0; hc < 4; ++hc) {
      FragA bk0, bk1;
#pragma unroll
      for (int j = 0; j < 8; ++j) {
        int k = hc * 32 + ((j >= 4) ? 16 : 0) + half * 8 + ((j & 3) << 1);
        bk0.u[j] = *(const unsigned int*)&Ks[lm * KS_STRIDE + k];
        bk1.u[j] = *(const unsigned int*)&Ks[(16 + lm) * KS_STRIDE + k];
      }
      s0.v = __builtin_amdgcn_wmma_f32_16x16x32_bf16(false, aq[hc].v, false, bk0.v, (short)0, s0.v, false, false);
      s1.v = __builtin_amdgcn_wmma_f32_16x16x32_bf16(false, aq[hc].v, false, bk1.v, (short)0, s1.v, false, false);
    }

    const int key0 = kb + lm;
    const int key1 = key0 + 16;
    float alpha[8];
#pragma unroll
    for (int r = 0; r < 8; ++r) {
      int qrow = qt * 16 + r + half * 8;
      float v0 = (key0 <= qrow) ? s0.f[r] * scale : -1e30f;
      float v1 = (key1 <= qrow) ? s1.f[r] * scale : -1e30f;
      float mr = fmaxf(v0, v1);
#pragma unroll
      for (int off = 1; off < 16; off <<= 1)
        mr = fmaxf(mr, __shfl_xor(mr, off, 32));
      float mn = fmaxf(m_i[r], mr);
      alpha[r] = __expf(m_i[r] - mn);
      float p0 = __expf(v0 - mn);
      float p1 = __expf(v1 - mn);
      float ps = p0 + p1;
#pragma unroll
      for (int off = 1; off < 16; off <<= 1)
        ps += __shfl_xor(ps, off, 32);
      l_i[r] = l_i[r] * alpha[r] + ps;
      m_i[r] = mn;
      s0.f[r] = p0; s1.f[r] = p1;
    }
#pragma unroll
    for (int f = 0; f < 8; ++f)
#pragma unroll
      for (int r = 0; r < 8; ++r) o[f].f[r] *= alpha[r];

    // P (C layout) -> LDS -> A-frag layout, bf16
#pragma unroll
    for (int r = 0; r < 8; ++r) {
      Ps[(r + half * 8) * 34 + lm]      = f2bf(s0.f[r]);
      Ps[(r + half * 8) * 34 + lm + 16] = f2bf(s1.f[r]);
    }
    __syncthreads();
    FragA ap;
#pragma unroll
    for (int j = 0; j < 8; ++j) {
      int k = ((j >= 4) ? 16 : 0) + half * 8 + ((j & 3) << 1);
      ap.u[j] = *(const unsigned int*)&Ps[lm * 34 + k];
    }

#pragma unroll
    for (int hf = 0; hf < 8; ++hf) {
      FragA bv;
#pragma unroll
      for (int j = 0; j < 8; ++j) {
        int k = ((j >= 4) ? 16 : 0) + half * 8 + ((j & 3) << 1);
        bv.u[j] = *(const unsigned int*)&Vs[(hf * 16 + lm) * VS_STRIDE + k];
      }
      o[hf].v = __builtin_amdgcn_wmma_f32_16x16x32_bf16(false, ap.v, false, bv.v, (short)0, o[hf].v, false, false);
    }
  }

#pragma unroll
  for (int r = 0; r < 8; ++r) l_i[r] = 1.0f / l_i[r];
#pragma unroll
  for (int hf = 0; hf < 8; ++hf)
#pragma unroll
    for (int r = 0; r < 8; ++r) {
      int row = qt * 16 + r + half * 8;
      long idx = (long)(b * T_ + row) * (NH_ * H_) + (long)n * H_ + hf * 16 + lm;
      Ab[idx] = f2bf(o[hf].f[r] * l_i[r]);
    }
}

// ---------------------------------------------------------------------------
extern "C" void kernel_launch(void* const* d_in, const int* in_sizes, int n_in,
                              void* d_out, int out_size, void* d_ws, size_t ws_size,
                              hipStream_t stream) {
  (void)in_sizes; (void)n_in; (void)out_size; (void)ws_size;

  const float* Xq   = (const float*)d_in[0];
  const float* Xkv  = (const float*)d_in[1];
  const int*   qpos = (const int*)d_in[2];
  const int*   kpos = (const int*)d_in[3];
  const float* Wq   = (const float*)d_in[4];
  const float* Wk   = (const float*)d_in[5];
  const float* Wv   = (const float*)d_in[6];
  const float* Wo   = (const float*)d_in[7];
  float* out = (float*)d_out;

  const int M = B_ * T_;                           // 4096 rows
  unsigned short* Qb = (unsigned short*)d_ws;      // (M, N*H)      16 MB
  unsigned short* Kb = Qb + (size_t)M * NH_ * H_;  // (M, K*H)       4 MB
  unsigned short* Vt = Kb + (size_t)M * KH_ * H_;  // (K*H, M) ^T    4 MB
  unsigned short* Ab = Vt + (size_t)M * KH_ * H_;  // (M, N*H)      16 MB

  dim3 blk(256);
  // Q = rope(Xq @ Wq) -> bf16
  gemm_bf16_wmma_kernel<<<dim3(M / 128, (NH_ * H_) / 128), blk, 0, stream>>>(
      Xq, nullptr, Wq, Qb, nullptr, qpos, D_, NH_ * H_, M, 1, 0);
  // K = rope(Xkv @ Wk) -> bf16
  gemm_bf16_wmma_kernel<<<dim3(M / 128, (KH_ * H_) / 128), blk, 0, stream>>>(
      Xkv, nullptr, Wk, Kb, nullptr, kpos, D_, KH_ * H_, M, 1, 0);
  // V^T = (Xkv @ Wv)^T -> bf16 column-major
  gemm_bf16_wmma_kernel<<<dim3(M / 128, (KH_ * H_) / 128), blk, 0, stream>>>(
      Xkv, nullptr, Wv, Vt, nullptr, nullptr, D_, KH_ * H_, M, 0, 1);
  // causal GQA flash attention (TDM-staged K/V) -> bf16
  flash_fwd_kernel<<<dim3(T_ / 16, B_ * NH_), dim3(32), 0, stream>>>(Qb, Kb, Vt, Ab);
  // out = attn @ Wo -> f32
  gemm_bf16_wmma_kernel<<<dim3(M / 128, D_ / 128), blk, 0, stream>>>(
      nullptr, Ab, Wo, nullptr, out, nullptr, NH_ * H_, D_, M, 0, 0);
}
